// TransformedConditionalAggregateKRR_70102456205629
// MI455X (gfx1250) — matrix operand order repaired and checked by reading, not compiled, not run on hardware
//
#include <hip/hip_runtime.h>

// CDNA5 / gfx1250 fused RBF-KRR kernel:
//   out[i] = softplus( sum_j exp(-0.5*||x_i - c_j||^2) * beta[j] )
// Dot products via v_wmma_f32_16x16x32_bf16; fp32 split into bf16 hi/lo
// (3 chained WMMAs per 16x16 tile) for ~fp32 accuracy.
//   exp(-0.5*||x-c||^2)*beta = exp(dot - 0.5*|x|^2) * (beta*exp(-0.5*|c|^2))
// so the column constant is pre-folded into a per-column weight w_j and the
// row constant is the (loop-invariant) WMMA C initializer: the per-element
// epilogue is exactly one v_exp_f32 + one fma.

typedef __attribute__((ext_vector_type(16))) __bf16          v16bf;
typedef __attribute__((ext_vector_type(8)))  float           v8f;
typedef __attribute__((ext_vector_type(8)))  unsigned short  v8us;

#define LOG2E_F 1.4426950408889634f

#if __has_builtin(__builtin_amdgcn_exp2f)
#define EXP2F(x) __builtin_amdgcn_exp2f(x)
#else
#define EXP2F(x) exp2f(x)
#endif

// cross-lane gather (wave32): value of `v` held by lane `srcLane`
static __device__ __forceinline__ float lane_get(int srcLane, float v) {
  return __int_as_float(__builtin_amdgcn_ds_bpermute(srcLane << 2, __float_as_int(v)));
}

static __device__ __forceinline__ unsigned short f32_to_bf16_rne(float f) {
  unsigned int u = __float_as_uint(f);
  u += 0x7FFFu + ((u >> 16) & 1u);   // round-to-nearest-even
  return (unsigned short)(u >> 16);
}
static __device__ __forceinline__ float bf16_to_f32(unsigned short h) {
  return __uint_as_float(((unsigned int)h) << 16);
}

union V16U { v16bf v; v8us u2[2]; unsigned short h[16]; };

#define WMMA_BF16(A, B, C) \
  __builtin_amdgcn_wmma_f32_16x16x32_bf16(false, (A), false, (B), (short)0, (C), false, false)

// ---------------------------------------------------------------------------
// Preprocess cov (M x 32 fp32) -> hl[M][64] = [hi bf16 x32 | lo bf16 x32]
// (128 B per row), and cw[M] = beta[j] * exp(-0.5*||c_j||^2).
// ---------------------------------------------------------------------------
__global__ __launch_bounds__(256) void krr_prep_kernel(
    const float* __restrict__ cov, const float* __restrict__ beta,
    unsigned short* __restrict__ hl, float* __restrict__ cw, int M)
{
  int j = blockIdx.x * blockDim.x + threadIdx.x;
  if (j >= M) return;
  const float* cr = cov + (size_t)j * 32;
  unsigned short* o = hl + (size_t)j * 64;
  float s = 0.0f;
#pragma unroll
  for (int i = 0; i < 32; ++i) {
    float v = cr[i];
    s += v * v;
    unsigned short h = f32_to_bf16_rne(v);
    o[i]      = h;
    o[32 + i] = f32_to_bf16_rne(v - bf16_to_f32(h));
  }
  cw[j] = EXP2F(-0.5f * LOG2E_F * s) * beta[j];
}

// A-operand builder: 16x32 bf16 hi/lo of x rows, pre-scaled by log2(e).
// rowc[k] = -0.5*log2e*||x_{rowBase+8g+k}||^2  (C/D layout row constants).
static __device__ __forceinline__ void build_a(
    const float* __restrict__ x, int N, int rowBase, int g, int l, int lane,
    V16U& ah, V16U& al, float rowc[8])
{
  float xs_half = 0.0f;
  int r = rowBase + l;
  if (r >= N) r = N - 1;
  const float* xr = x + (size_t)r * 32;
  const int base = g << 4;  // A layout: lane(g,l) -> row M=l, K = base..base+15
#pragma unroll
  for (int i = 0; i < 8; ++i) {
    float r0 = xr[base + i];
    float r1 = xr[base + 8 + i];
    xs_half += r0 * r0 + r1 * r1;
    float s0 = r0 * LOG2E_F, s1 = r1 * LOG2E_F;
    unsigned short h0 = f32_to_bf16_rne(s0);
    unsigned short h1 = f32_to_bf16_rne(s1);
    ah.h[i]     = h0; al.h[i]     = f32_to_bf16_rne(s0 - bf16_to_f32(h0));
    ah.h[8 + i] = h1; al.h[8 + i] = f32_to_bf16_rne(s1 - bf16_to_f32(h1));
  }
  const float xsq    = xs_half + lane_get(lane ^ 16, xs_half);
  const float myRowC = -0.5f * LOG2E_F * xsq;
#pragma unroll
  for (int k = 0; k < 8; ++k) rowc[k] = lane_get(k + (g << 3), myRowC);
}

// Column reduction across lanes/waves + softplus + store.
static __device__ __forceinline__ void reduce_store(
    float accOut[8], float* __restrict__ out, int N, int rowBase,
    int w, int g, int l, int lane)
{
  __shared__ float red[8][16];
#pragma unroll
  for (int m = 1; m < 16; m <<= 1) {
#pragma unroll
    for (int k = 0; k < 8; ++k) accOut[k] += lane_get(lane ^ m, accOut[k]);
  }
  if (l == 0) {
#pragma unroll
    for (int k = 0; k < 8; ++k) red[w][(g << 3) + k] = accOut[k];
  }
  __syncthreads();
  if (threadIdx.x < 16) {
    float s = 0.0f;
#pragma unroll
    for (int ww = 0; ww < 8; ++ww) s += red[ww][threadIdx.x];
    int row = rowBase + (int)threadIdx.x;
    if (row < N) out[row] = fmaxf(s, 0.0f) + log1pf(__expf(-fabsf(s)));
  }
}

// ---------------------------------------------------------------------------
// Main fused kernel (preprocessed B). Each block owns 16 x-rows; 8 waves
// split the column tiles. Full tiles run guard-free with scalar loop bounds.
// ---------------------------------------------------------------------------
__global__ __launch_bounds__(256) void krr_fused_pre(
    const float* __restrict__ x,
    const unsigned short* __restrict__ hl, const float* __restrict__ cw,
    float* __restrict__ out, int N, int M)
{
  const int lane = threadIdx.x & 31;
  const int w    = __builtin_amdgcn_readfirstlane(threadIdx.x >> 5); // scalar wave id
  const int g    = lane >> 4;
  const int l    = lane & 15;
  const int rowBase = blockIdx.x << 4;

  V16U ah, al;
  float rowc[8];
  build_a(x, N, rowBase, g, l, lane, ah, al, rowc);
  v8f rowcv;   // loop-invariant WMMA C initializer
#pragma unroll
  for (int k = 0; k < 8; ++k) rowcv[k] = rowc[k];

  float accOut[8] = {0.f,0.f,0.f,0.f,0.f,0.f,0.f,0.f};

  const int nFull = M >> 4;  // guard-free full 16-column tiles
  // B layout (16-bit, 32x16): lane(g,l) holds col N=l, K = g*16..g*16+15.
  // byte offset into hl for tile jt: (jt*16 + l)*128 + g*32
  unsigned off = (unsigned)(((w << 4) + l) * 128 + (g << 5));
  int      cji = (w << 4) + l;          // index into cw[]
  const char* hlb = (const char*)hl;

#pragma unroll 2
  for (int jt = w; jt < nFull; jt += 8) {
    V16U bh, bl;
    bh.u2[0] = *(const v8us*)(hlb + off);
    bh.u2[1] = *(const v8us*)(hlb + off + 16);
    bl.u2[0] = *(const v8us*)(hlb + off + 64);
    bl.u2[1] = *(const v8us*)(hlb + off + 80);
    float wv = cw[cji];

    // acc = log2e*(x.c) - 0.5*log2e*|x|^2 ; column factor lives in wv
    v8f acc = rowcv;
    acc = WMMA_BF16(al.v, bh.v, acc);
    acc = WMMA_BF16(ah.v, bl.v, acc);
    acc = WMMA_BF16(ah.v, bh.v, acc);

#pragma unroll
    for (int k = 0; k < 8; ++k)
      accOut[k] = fmaf(EXP2F(acc[k]), wv, accOut[k]);

    off += 8 * 16 * 128;   // advance 8 tiles
    cji += 8 * 16;
  }

  // tail tile (M % 16 != 0), owned by wave (nFull & 7)
  if ((M & 15) && w == (nFull & 7)) {
    int  cj    = (nFull << 4) + l;
    bool valid = (cj < M);
    int  cjc   = valid ? cj : (M - 1);
    const unsigned short* rowp = hl + (size_t)cjc * 64 + (g << 4);
    V16U bh, bl;
    bh.u2[0] = *(const v8us*)(rowp);
    bh.u2[1] = *(const v8us*)(rowp + 8);
    bl.u2[0] = *(const v8us*)(rowp + 32);
    bl.u2[1] = *(const v8us*)(rowp + 40);
    float wv = valid ? cw[cjc] : 0.0f;
    v8f acc = rowcv;
    acc = WMMA_BF16(al.v, bh.v, acc);
    acc = WMMA_BF16(ah.v, bl.v, acc);
    acc = WMMA_BF16(ah.v, bh.v, acc);
#pragma unroll
    for (int k = 0; k < 8; ++k)
      accOut[k] = fmaf(EXP2F(acc[k]), wv, accOut[k]);
  }

  reduce_store(accOut, out, N, rowBase, w, g, l, lane);
}

// ---------------------------------------------------------------------------
// Fallback (no workspace): in-loop B conversion, guarded loop.
// ---------------------------------------------------------------------------
__global__ __launch_bounds__(256) void krr_fused_fallback(
    const float* __restrict__ x, const float* __restrict__ cov,
    const float* __restrict__ beta, float* __restrict__ out, int N, int M)
{
  const int lane = threadIdx.x & 31;
  const int w    = __builtin_amdgcn_readfirstlane(threadIdx.x >> 5);
  const int g    = lane >> 4;
  const int l    = lane & 15;
  const int rowBase = blockIdx.x << 4;

  V16U ah, al;
  float rowc[8];
  build_a(x, N, rowBase, g, l, lane, ah, al, rowc);

  float accOut[8] = {0.f,0.f,0.f,0.f,0.f,0.f,0.f,0.f};
  const int nColTiles = (M + 15) >> 4;
  for (int jt = w; jt < nColTiles; jt += 8) {
    int  cj    = (jt << 4) + l;
    bool valid = (cj < M);
    int  cjc   = valid ? cj : (M - 1);
    const float* cr = cov + (size_t)cjc * 32 + (g << 4);
    V16U bh, bl;
    float cs_half = 0.0f;
#pragma unroll
    for (int i = 0; i < 16; ++i) {
      float rv = cr[i];
      cs_half += rv * rv;
      unsigned short h0 = f32_to_bf16_rne(rv);
      bh.h[i] = h0;
      bl.h[i] = f32_to_bf16_rne(rv - bf16_to_f32(h0));
    }
    const float csq  = cs_half + lane_get(lane ^ 16, cs_half);
    const float colc = -0.5f * LOG2E_F * csq;
    const float bet  = valid ? beta[cj] : 0.0f;
    v8f acc;
#pragma unroll
    for (int k = 0; k < 8; ++k) acc[k] = rowc[k] + colc;
    acc = WMMA_BF16(al.v, bh.v, acc);
    acc = WMMA_BF16(ah.v, bl.v, acc);
    acc = WMMA_BF16(ah.v, bh.v, acc);
#pragma unroll
    for (int k = 0; k < 8; ++k)
      accOut[k] = fmaf(EXP2F(acc[k]), bet, accOut[k]);
  }

  reduce_store(accOut, out, N, rowBase, w, g, l, lane);
}

extern "C" void kernel_launch(void* const* d_in, const int* in_sizes, int n_in,
                              void* d_out, int out_size, void* d_ws, size_t ws_size,
                              hipStream_t stream) {
  const float* x    = (const float*)d_in[0];   // (N, 32) fp32
  const float* cov  = (const float*)d_in[1];   // (M, 32) fp32
  const float* beta = (const float*)d_in[2];   // (M,)    fp32
  float* out = (float*)d_out;                  // (N,)    fp32
  int N = in_sizes[0] / 32;
  int M = in_sizes[2];
  int nRowTiles = (N + 15) / 16;

  const size_t hlBytes = (size_t)M * 64 * sizeof(unsigned short); // 128 B/row
  const size_t cwBytes = (size_t)M * sizeof(float);

  if (ws_size >= hlBytes + cwBytes) {
    unsigned short* hl = (unsigned short*)d_ws;
    float*          cw = (float*)((char*)d_ws + hlBytes);
    krr_prep_kernel<<<(M + 255) / 256, 256, 0, stream>>>(cov, beta, hl, cw, M);
    krr_fused_pre<<<nRowTiles, 256, 0, stream>>>(x, hl, cw, out, N, M);
  } else {
    krr_fused_fallback<<<nRowTiles, 256, 0, stream>>>(x, cov, beta, out, N, M);
  }
  (void)n_in; (void)out_size;
}
